// AttentionLayer_80152679678710
// MI455X (gfx1250) — matrix operand order repaired and checked by reading, not compiled, not run on hardware
//
#include <hip/hip_runtime.h>
#include <hip/hip_bf16.h>
#include <math.h>

// ---------------------------------------------------------------------------
// CDNA5 / gfx1250 bf16 WMMA attention layer.
// GEMMs: v_wmma_f32_16x16x32_bf16. Attention K/V tiles: TDM tensor_load_to_lds.
// ---------------------------------------------------------------------------

typedef __attribute__((ext_vector_type(16))) __bf16 v16bf;
typedef __attribute__((ext_vector_type(8)))  float  v8f;
typedef unsigned int uint32x4 __attribute__((ext_vector_type(4)));
typedef int          int32x8  __attribute__((ext_vector_type(8)));
typedef int          int32x4  __attribute__((ext_vector_type(4)));

union ABu { v16bf v; unsigned short u[16]; };

__device__ __forceinline__ unsigned short f2bf(float f) {
  unsigned int x = __float_as_uint(f);
  return (unsigned short)((x + 0x7FFFu + ((x >> 16) & 1u)) >> 16);
}

// A-matrix fragment (16xK, 16-bit): lane&15 = row; lane half selects K
// subchunks {0..7,16..23} vs {8..15,24..31}. Two 16-byte loads.
__device__ __forceinline__ v16bf load_a_bf(const unsigned short* base, int stride, int lane) {
  int half = lane >> 4, r = lane & 15;
  const unsigned short* p = base + (size_t)r * stride + half * 8;
  ABu f;
  *(uint4*)&f.u[0] = *(const uint4*)(p);
  *(uint4*)&f.u[8] = *(const uint4*)(p + 16);
  return f.v;
}

// B-matrix fragment (Kx16, 16-bit): lane&15 = column; lane half selects K
// subchunks {0..7,16..23} vs {8..15,24..31}. Column c contiguous along K
// with 'stride' elements between columns.
__device__ __forceinline__ v16bf load_b_bf(const unsigned short* base, int stride, int lane) {
  int half = lane >> 4, c = lane & 15;
  const unsigned short* p = base + (size_t)c * stride + half * 8;
  ABu f;
  *(uint4*)&f.u[0] = *(const uint4*)(p);
  *(uint4*)&f.u[8] = *(const uint4*)(p + 16);
  return f.v;
}

__device__ __forceinline__ v8f wmma_bf16(v16bf a, v16bf b, v8f c) {
  return __builtin_amdgcn_wmma_f32_16x16x32_bf16(false, a, false, b, (short)0, c, false, false);
}

// ---------------------------------------------------------------------------
// TDM: 2D tile load Global -> LDS. Builds D# group0 (128b) + group1 (256b)
// per CDNA5 ISA ch.8; remaining groups zero (2D tensor). data_size = 2 bytes.
// tensor dims == tile dims so the OOB path never triggers.
// ---------------------------------------------------------------------------
__device__ __forceinline__ void tdm_load_2d_bf16(unsigned lds_addr, const void* gaddr,
                                                 unsigned tile_d0, unsigned tile_d1,
                                                 unsigned stride0_elems) {
  unsigned long long ga = (unsigned long long)(size_t)gaddr;
  uint32x4 g0;
  g0[0] = 1u;                                            // count=1 (valid), user mode
  g0[1] = lds_addr;                                      // LDS byte address
  g0[2] = (unsigned)(ga & 0xFFFFFFFFu);                  // global_addr[31:0]
  g0[3] = (unsigned)((ga >> 32) & 0x01FFFFFFu) | (2u << 30);  // addr[56:32] | type=2
  int32x8 g1;
  g1[0] = (int)(1u << 16);                               // wg_mask=0, data_size=1 (2B)
  g1[1] = (int)((tile_d0 & 0xFFFFu) << 16);              // tensor_dim0[15:0]
  g1[2] = (int)((tile_d0 >> 16) | ((tile_d1 & 0xFFFFu) << 16));   // dim0 hi | dim1 lo
  g1[3] = (int)((tile_d1 >> 16) | (tile_d0 << 16));      // dim1 hi | tile_dim0
  g1[4] = (int)tile_d1;                                  // tile_dim1 | tile_dim2=0
  g1[5] = (int)stride0_elems;                            // tensor_dim0_stride[31:0]
  g1[6] = 0;                                             // stride0 hi | stride1 lo
  g1[7] = 0;
  int32x4 z4 = {0, 0, 0, 0};
  int32x8 z8 = {0, 0, 0, 0, 0, 0, 0, 0};
  __builtin_amdgcn_tensor_load_to_lds(g0, g1, z4, z4, z8, 0);
}

// ---------------------------------------------------------------------------
// Stage 1: f32 -> bf16 conversion (weights and q/k/v activations)
// ---------------------------------------------------------------------------
__global__ __launch_bounds__(256) void cvt_bf16_kernel(const float* __restrict__ src,
                                                       unsigned short* __restrict__ dst, int n) {
  int i = blockIdx.x * 256 + threadIdx.x;
  if (i < n) dst[i] = f2bf(src[i]);
}

// ---------------------------------------------------------------------------
// Stage 2: past_k/past_v -> f32 cat outputs + bf16 caches (V transposed/head)
// ---------------------------------------------------------------------------
__global__ __launch_bounds__(256) void copy_past_kernel(
    const float* __restrict__ pk, const float* __restrict__ pv,
    float* __restrict__ kcatF, float* __restrict__ vcatF,
    unsigned short* __restrict__ kcatB, unsigned short* __restrict__ vTB) {
  int i = blockIdx.x * 256 + threadIdx.x;
  if (i >= 2 * 2048 * 1024) return;
  int n = i & 1023;
  int t = i >> 10;
  int s = t & 2047;
  int b = t >> 11;
  float kval = pk[i];
  float vval = pv[i];
  size_t o = (size_t)(b * 4096 + s) * 1024 + n;
  kcatF[o] = kval;
  kcatB[o] = f2bf(kval);
  vcatF[o] = vval;
  int h = n >> 6, dl = n & 63;
  vTB[(size_t)((b * 16 + h) * 64 + dl) * 4096 + s] = f2bf(vval);
}

// ---------------------------------------------------------------------------
// Stage 3/5: GEMM  Y[m,n] = sum_k A[m,k]*W[n,k] + bias[n], A bf16 [4096][1024]
// One wave -> 16x64 tile, K-loop step 32, 4 WMMAs/iter.
// mode 0: out bf16 qp (flat)
// mode 1: out f32 k_cat half + bf16 kcat cache
// mode 2: out f32 v_cat half + bf16 vT cache (per-head transposed)
// mode 3: out f32 x (flat)
// ---------------------------------------------------------------------------
__global__ __launch_bounds__(256) void gemm16_kernel(
    const unsigned short* __restrict__ Ab, const unsigned short* __restrict__ Wb,
    const float* __restrict__ bias, float* __restrict__ outF,
    unsigned short* __restrict__ outB, int mode) {
  const int M = 4096, N = 1024, K = 1024;
  int lane = threadIdx.x & 31;
  int wave = blockIdx.x * (blockDim.x >> 5) + (threadIdx.x >> 5);
  const int nTilesN = N / 64;  // 16
  int tm = wave / nTilesN;
  int tn = wave % nTilesN;
  if (tm >= M / 16) return;
  int row0 = tm * 16;
  int n0 = tn * 64;
  int half = lane >> 4, lr = lane & 15;

  v8f acc0 = {}, acc1 = {}, acc2 = {}, acc3 = {};

  for (int k0 = 0; k0 < K; k0 += 32) {
    v16bf a = load_a_bf(Ab + (size_t)row0 * K + k0, K, lane);
    v16bf b0 = load_b_bf(Wb + (size_t)(n0 + 0)  * K + k0, K, lane);
    v16bf b1 = load_b_bf(Wb + (size_t)(n0 + 16) * K + k0, K, lane);
    v16bf b2 = load_b_bf(Wb + (size_t)(n0 + 32) * K + k0, K, lane);
    v16bf b3 = load_b_bf(Wb + (size_t)(n0 + 48) * K + k0, K, lane);
    acc0 = wmma_bf16(a, b0, acc0);
    acc1 = wmma_bf16(a, b1, acc1);
    acc2 = wmma_bf16(a, b2, acc2);
    acc3 = wmma_bf16(a, b3, acc3);
  }

  v8f accs[4] = {acc0, acc1, acc2, acc3};
#pragma unroll
  for (int g = 0; g < 4; ++g) {
    int n = n0 + g * 16 + lr;
    float bv = bias[n];
#pragma unroll
    for (int e = 0; e < 8; ++e) {
      int m = row0 + e + half * 8;
      int b = m >> 11, s = m & 2047;
      float val = accs[g][e] + bv;
      if (mode == 1 || mode == 2) outF[(size_t)(b * 4096 + 2048 + s) * N + n] = val;
      if (mode == 3)              outF[(size_t)m * N + n] = val;
      if (mode == 0)              outB[(size_t)m * N + n] = f2bf(val);
      if (mode == 1)              outB[(size_t)(b * 4096 + 2048 + s) * N + n] = f2bf(val);
      if (mode == 2) {
        int h = n >> 6, dl = n & 63;
        outB[(size_t)((b * 16 + h) * 64 + dl) * 4096 + 2048 + s] = f2bf(val);
      }
    }
  }
}

// ---------------------------------------------------------------------------
// Stage 4: flash attention, one wave per (b, h, 16-row q tile), 4 waves/block.
// K tiles (32x64 bf16, row stride 1024) TDM-loaded to LDS, double-buffered.
// V tiles (64x32 bf16 from per-head-transposed cache, row stride 4096) TDM'd.
// Per 32-kv chunk: 4 WMMAs for S, online softmax, P->LDS transpose, 4 WMMAs P@V.
// ---------------------------------------------------------------------------
__global__ __launch_bounds__(128) void attn_kernel(
    const unsigned short* __restrict__ qp,   // [B*2048][1024] bf16
    const unsigned short* __restrict__ kc,   // [B*4096][1024] bf16
    const unsigned short* __restrict__ vT,   // [B][H][64][4096] bf16
    const unsigned char* __restrict__ mask,  // [B][2048][4096] bool
    unsigned short* __restrict__ attnB) {    // [B*2048][1024] bf16
  // per wave: K ping-pong 2*2048, V 2048, P 640 ushorts  (13568 B)
  __shared__ __align__(16) unsigned short smem[4][6784];

  int lane = threadIdx.x & 31;
  int wid = threadIdx.x >> 5;
  int wave = blockIdx.x * 4 + wid;  // 0..4095
  int qt = wave & 127;
  int bh = wave >> 7;
  int h = bh & 15;
  int b = bh >> 4;
  int half = lane >> 4, lr = lane & 15, rbase = half * 8;

  unsigned short* Kt[2];
  Kt[0] = &smem[wid][0];
  Kt[1] = &smem[wid][2048];
  unsigned short* Vt = &smem[wid][4096];
  unsigned short* P  = &smem[wid][6144];
  unsigned ldsK[2], ldsV;
  ldsK[0] = (unsigned)(size_t)(void*)Kt[0];
  ldsK[1] = (unsigned)(size_t)(void*)Kt[1];
  ldsV    = (unsigned)(size_t)(void*)Vt;

  const unsigned short* qbase = qp + (size_t)(b * 2048 + qt * 16) * 1024 + h * 64;
  v16bf aq0 = load_a_bf(qbase, 1024, lane);       // d 0..31
  v16bf aq1 = load_a_bf(qbase + 32, 1024, lane);  // d 32..63

  const unsigned char* mbase = mask + (size_t)(b * 2048 + qt * 16) * 4096;
  const unsigned short* kbase = kc + (size_t)(b * 4096) * 1024 + h * 64;
  const unsigned short* vbase = vT + (size_t)((b * 16 + h) * 64) * 4096;

  float rmax[8], rsum[8];
#pragma unroll
  for (int e = 0; e < 8; ++e) { rmax[e] = -INFINITY; rsum[e] = 0.0f; }
  v8f O0 = {}, O1 = {}, O2 = {}, O3 = {};

  // prologue: DMA first K and V tiles
  tdm_load_2d_bf16(ldsK[0], kbase, 64, 32, 1024);
  tdm_load_2d_bf16(ldsV, vbase, 32, 64, 4096);

  for (int kv0 = 0; kv0 < 4096; kv0 += 32) {
    int cur = (kv0 >> 5) & 1;
    __builtin_amdgcn_s_wait_tensorcnt(0);
    asm volatile("" ::: "memory");
    // prefetch next K tile into the other buffer while we compute
    if (kv0 + 32 < 4096)
      tdm_load_2d_bf16(ldsK[cur ^ 1], kbase + (size_t)(kv0 + 32) * 1024, 64, 32, 1024);

    // ---- S = Q K^T (16 q-rows x 32 kv-cols), K tile from LDS ------------
    v8f s0 = {}, s1 = {};
    {
      v16bf b00 = load_b_bf(Kt[cur], 64, lane);            // kv rows 0..15, d 0..31
      v16bf b01 = load_b_bf(Kt[cur] + 32, 64, lane);       // d 32..63
      s0 = wmma_bf16(aq0, b00, s0);
      s0 = wmma_bf16(aq1, b01, s0);
      v16bf b10 = load_b_bf(Kt[cur] + 16 * 64, 64, lane);  // kv rows 16..31
      v16bf b11 = load_b_bf(Kt[cur] + 16 * 64 + 32, 64, lane);
      s1 = wmma_bf16(aq0, b10, s1);
      s1 = wmma_bf16(aq1, b11, s1);
    }
    // ---- scale + mask + online softmax ----------------------------------
    float corr[8];
#pragma unroll
    for (int e = 0; e < 8; ++e) {
      int r = e + rbase;  // q row within tile
      const unsigned char* mp = mbase + (size_t)r * 4096 + kv0 + lr;
      float s0e = s0[e] * 0.125f - (mp[0] ? 10000.0f : 0.0f);
      float s1e = s1[e] * 0.125f - (mp[16] ? 10000.0f : 0.0f);
      float mc = fmaxf(s0e, s1e);
      mc = fmaxf(mc, __shfl_xor(mc, 1));
      mc = fmaxf(mc, __shfl_xor(mc, 2));
      mc = fmaxf(mc, __shfl_xor(mc, 4));
      mc = fmaxf(mc, __shfl_xor(mc, 8));
      float mn = fmaxf(rmax[e], mc);
      float cf = __expf(rmax[e] - mn);
      float p0 = __expf(s0e - mn);
      float p1 = __expf(s1e - mn);
      float rs = p0 + p1;
      rs += __shfl_xor(rs, 1);
      rs += __shfl_xor(rs, 2);
      rs += __shfl_xor(rs, 4);
      rs += __shfl_xor(rs, 8);
      rsum[e] = rsum[e] * cf + rs;
      rmax[e] = mn;
      corr[e] = cf;
      P[r * 40 + lr] = f2bf(p0);
      P[r * 40 + lr + 16] = f2bf(p1);
    }
    // same-wave LDS RAW: P written above, re-read transposed as A-fragment
    asm volatile("s_wait_dscnt 0" ::: "memory");
    v16bf ap;
    {
      const unsigned short* p = P + (size_t)lr * 40 + half * 8;
      ABu f;
      *(uint4*)&f.u[0] = *(const uint4*)(p);
      *(uint4*)&f.u[8] = *(const uint4*)(p + 16);
      ap = f.v;
    }
    // ---- O = O*corr + P @ V_chunk, V tile from LDS ----------------------
#pragma unroll
    for (int e = 0; e < 8; ++e) {
      O0[e] = O0[e] * corr[e];
      O1[e] = O1[e] * corr[e];
      O2[e] = O2[e] * corr[e];
      O3[e] = O3[e] * corr[e];
    }
    v16bf bv0 = load_b_bf(Vt + 0  * 32, 32, lane);
    v16bf bv1 = load_b_bf(Vt + 16 * 32, 32, lane);
    v16bf bv2 = load_b_bf(Vt + 32 * 32, 32, lane);
    v16bf bv3 = load_b_bf(Vt + 48 * 32, 32, lane);
    O0 = wmma_bf16(ap, bv0, O0);
    O1 = wmma_bf16(ap, bv1, O1);
    O2 = wmma_bf16(ap, bv2, O2);
    O3 = wmma_bf16(ap, bv3, O3);
    asm volatile("" ::: "memory");
    // V frags are in VGPRs now; safe to overwrite the V tile
    if (kv0 + 32 < 4096)
      tdm_load_2d_bf16(ldsV, vbase + kv0 + 32, 32, 64, 4096);
  }

  // ---- normalize and store bf16 -----------------------------------------
  float rl[8];
#pragma unroll
  for (int e = 0; e < 8; ++e) rl[e] = 1.0f / rsum[e];
  v8f Os[4] = {O0, O1, O2, O3};
#pragma unroll
  for (int g = 0; g < 4; ++g) {
#pragma unroll
    for (int e = 0; e < 8; ++e) {
      int row = b * 2048 + qt * 16 + e + rbase;
      attnB[(size_t)row * 1024 + h * 64 + g * 16 + lr] = f2bf(Os[g][e] * rl[e]);
    }
  }
}

// ---------------------------------------------------------------------------
// Launch
// ---------------------------------------------------------------------------
extern "C" void kernel_launch(void* const* d_in, const int* in_sizes, int n_in,
                              void* d_out, int out_size, void* d_ws, size_t ws_size,
                              hipStream_t stream) {
  const float* q      = (const float*)d_in[0];
  const float* k      = (const float*)d_in[1];
  const float* v      = (const float*)d_in[2];
  const float* past_k = (const float*)d_in[3];
  const float* past_v = (const float*)d_in[4];
  const unsigned char* mask = (const unsigned char*)d_in[5];
  const float* Wq = (const float*)d_in[6];
  const float* bq = (const float*)d_in[7];
  const float* Wk = (const float*)d_in[8];
  const float* bk = (const float*)d_in[9];
  const float* Wv = (const float*)d_in[10];
  const float* bv = (const float*)d_in[11];
  const float* Wl = (const float*)d_in[12];
  const float* bl = (const float*)d_in[13];

  float* out   = (float*)d_out;
  float* xO    = out;                   // (B,2048,1024)
  float* kcatF = out + 4194304;         // (B,4096,1024)
  float* vcatF = out + 12582912;        // (B,4096,1024)

  char* ws = (char*)d_ws;
  unsigned short* q_b    = (unsigned short*)(ws + 0);         //  8 MB
  unsigned short* k_b    = (unsigned short*)(ws + 8388608);   //  8 MB
  unsigned short* v_b    = (unsigned short*)(ws + 16777216);  //  8 MB
  unsigned short* qp_b   = (unsigned short*)(ws + 25165824);  //  8 MB
  unsigned short* kcat_b = (unsigned short*)(ws + 33554432);  // 16 MB
  unsigned short* vT_b   = (unsigned short*)(ws + 50331648);  // 16 MB
  unsigned short* attn_b = (unsigned short*)(ws + 67108864);  //  8 MB
  unsigned short* Wq_b   = (unsigned short*)(ws + 75497472);  //  2 MB
  unsigned short* Wk_b   = (unsigned short*)(ws + 77594624);
  unsigned short* Wv_b   = (unsigned short*)(ws + 79691776);
  unsigned short* Wl_b   = (unsigned short*)(ws + 81788928);

  // 1) bf16 conversions (weights + activations)
  cvt_bf16_kernel<<<4096, 256, 0, stream>>>(Wq, Wq_b, 1048576);
  cvt_bf16_kernel<<<4096, 256, 0, stream>>>(Wk, Wk_b, 1048576);
  cvt_bf16_kernel<<<4096, 256, 0, stream>>>(Wv, Wv_b, 1048576);
  cvt_bf16_kernel<<<4096, 256, 0, stream>>>(Wl, Wl_b, 1048576);
  cvt_bf16_kernel<<<16384, 256, 0, stream>>>(q, q_b, 4194304);
  cvt_bf16_kernel<<<16384, 256, 0, stream>>>(k, k_b, 4194304);
  cvt_bf16_kernel<<<16384, 256, 0, stream>>>(v, v_b, 4194304);

  // 2) past kv -> f32 outputs + bf16 caches
  copy_past_kernel<<<16384, 256, 0, stream>>>(past_k, past_v, kcatF, vcatF, kcat_b, vT_b);

  // 3) projections (4096 waves -> 512 blocks x 8 waves)
  gemm16_kernel<<<512, 256, 0, stream>>>(q_b, Wq_b, bq, nullptr, qp_b, 0);
  gemm16_kernel<<<512, 256, 0, stream>>>(k_b, Wk_b, bk, kcatF, kcat_b, 1);
  gemm16_kernel<<<512, 256, 0, stream>>>(v_b, Wv_b, bv, vcatF, vT_b, 2);

  // 4) flash attention (4096 waves -> 1024 blocks x 4 waves)
  attn_kernel<<<1024, 128, 0, stream>>>(qp_b, kcat_b, vT_b, mask, attn_b);

  // 5) output projection -> x
  gemm16_kernel<<<512, 256, 0, stream>>>(attn_b, Wl_b, bl, xO, nullptr, 3);
}
